// Critic_71244917506250
// MI455X (gfx1250) — matrix operand order repaired and checked by reading, not compiled, not run on hardware
//
#include <hip/hip_runtime.h>
#include <hip/hip_bf16.h>
#include <stdint.h>

// ---------------------------------------------------------------------------
// CDNA5 / gfx1250 GNN critic. All GEMMs via v_wmma_f32_16x16x32_bf16
// (bf16 operands, fp32 accumulate). Edge gather tiles are staged to LDS with
// global_load_async_to_lds_b128 (ASYNCcnt path). Scatter-mean uses
// global_atomic_add_f32 into an L2-resident [N,256] f32 mailbox (102MB < 192MB L2).
// Full 16-row tiles take a uniform (SGPR-branch) unguarded fast path.
// ---------------------------------------------------------------------------

#define USE_ASYNC_LDS 1

typedef unsigned short u16;
typedef __bf16  bf16x16 __attribute__((ext_vector_type(16)));
typedef float   f32x8   __attribute__((ext_vector_type(8)));
typedef u16     u16x16  __attribute__((ext_vector_type(16)));
typedef u16     u16x8   __attribute__((ext_vector_type(8)));

__device__ __forceinline__ u16 f2bf(float f) {
  union { float f; unsigned u; } v; v.f = f;
  unsigned r = v.u + 0x7FFFu + ((v.u >> 16) & 1u);   // round-to-nearest-even
  return (u16)(r >> 16);
}

// ---------------------------------------------------------------------------
// WMMA fragment layout helpers (ISA 7.12.2, wave32):
//   A (16x32 bf16): lane L -> M = L&15 ; k-half kb = (L>=16)?8:0
//                   elems 0..7  -> k = kb+0..7 ; elems 8..15 -> k = kb+16..23
//   B (32x16 bf16): lane L -> N = L&15 ; elems 0..7 -> k = kb+0..7 ; elems 8..15 -> k = kb+16..23
//   C/D (16x16 f32): vgpr r, lane L -> M = r + ((L>=16)?8:0), N = L&15
// Weights are pre-packed so each lane's B fragment is 16 contiguous ushorts.
// ---------------------------------------------------------------------------

__device__ __forceinline__ bf16x16 load_a_frag_lds(const u16* rowp) {
  u16x8 lo = *(const u16x8*)(rowp);
  u16x8 hi = *(const u16x8*)(rowp + 16);
  u16x16 au = __builtin_shufflevector(lo, hi, 0,1,2,3,4,5,6,7,8,9,10,11,12,13,14,15);
  return __builtin_bit_cast(bf16x16, au);
}

__device__ __forceinline__ bf16x16 load_b_frag(const u16* p) {
  u16x16 bu = *(const u16x16*)p;          // 32B contiguous -> 2x global_load_b128
  return __builtin_bit_cast(bf16x16, bu);
}

// ---------------------------------------------------------------------------
// Weight pre-pack:  dst[(((l*ksteps + s)*16 + b)*32 + lane)*16 + e] =
//                   bf16( src[ (l*256 + b*16 + (lane&15)) * kstride
//                              + s*32 + ((lane>=16)?8:0) + (e&7) + ((e>=8)?16:0) ] )
// ---------------------------------------------------------------------------
__global__ void pack_b_kernel(const float* __restrict__ src, u16* __restrict__ dst,
                              int layers, int ksteps, int kstride) {
  size_t total = (size_t)layers * ksteps * 16 * 32 * 16;
  for (size_t idx = (size_t)blockIdx.x * blockDim.x + threadIdx.x; idx < total;
       idx += (size_t)gridDim.x * blockDim.x) {
    int e = (int)(idx & 15);
    size_t r = idx >> 4;
    int lane = (int)(r & 31); r >>= 5;
    int b = (int)(r & 15);    r >>= 4;
    int s = (int)(r % (size_t)ksteps);
    int l = (int)(r / (size_t)ksteps);
    int n = b * 16 + (lane & 15);
    int k = s * 32 + ((lane >> 4) << 3) + (e & 7) + ((e >> 3) << 4);
    dst[idx] = f2bf(src[((size_t)l * 256 + n) * kstride + k]);
  }
}

__global__ void init_h_kernel(const int* __restrict__ gate, const float* __restrict__ emb,
                              u16* __restrict__ h_bf, int N) {
  size_t idx = (size_t)blockIdx.x * blockDim.x + threadIdx.x;
  size_t total = (size_t)N * 256;
  if (idx < total) {
    int n = (int)(idx >> 8), c = (int)(idx & 255);
    h_bf[idx] = f2bf(emb[(size_t)gate[n] * 256 + c]);
  }
}

__global__ void zero_f32_kernel(float* __restrict__ p, size_t n) {
  for (size_t i = (size_t)blockIdx.x * blockDim.x + threadIdx.x; i < n;
       i += (size_t)gridDim.x * blockDim.x) p[i] = 0.0f;
}

__global__ void degree_kernel(const int* __restrict__ dst, float* __restrict__ deg, int E) {
  int e = blockIdx.x * blockDim.x + threadIdx.x;
  if (e < E) atomicAdd(&deg[dst[e]], 1.0f);
}

__global__ void invdeg_kernel(const float* __restrict__ deg, float* __restrict__ invd, int N) {
  int n = blockIdx.x * blockDim.x + threadIdx.x;
  if (n < N) invd[n] = 1.0f / fmaxf(deg[n], 1.0f);
}

// ---------------------------------------------------------------------------
// Edge kernel: tmp = leaky_relu([h[src], w] @ W1^T) ; hN[dst] += tmp
// One WG (8 waves) per 16-edge tile; wave handles 2 of 16 column blocks.
// ---------------------------------------------------------------------------
__launch_bounds__(256)
__global__ void edge_gemm_scatter(const u16* __restrict__ h_bf,
                                  const int* __restrict__ src,
                                  const int* __restrict__ dst,
                                  const float* __restrict__ w,
                                  const float* __restrict__ W1raw,  // [256][259] (this layer)
                                  const u16*  __restrict__ W1pack,  // [8][16][32][16] (this layer)
                                  float* __restrict__ hN, int E) {
  __shared__ __align__(32) u16 sA[16 * 256];   // 16 edges x 256 bf16
  __shared__ float sW[16 * 3];
  __shared__ int   sDst[16];

  const int t    = threadIdx.x;
  const int e0   = blockIdx.x * 16;
  const bool full = (e0 + 16 <= E);            // uniform: tail tile only when E%16!=0

  if (t < 16) { int e = e0 + t; sDst[t] = (e < E) ? dst[e] : 0; }
  if (t >= 16 && t < 64) {
    int q = t - 16;                            // 0..47 -> 16 edges x 3 feats
    int e = e0 + q / 3;
    sW[q] = (e < E) ? w[(size_t)e * 3 + (q % 3)] : 0.0f;
  }
  {
    const int row = t >> 4;
    const int c   = (t & 15) * 16;
    int e = e0 + row;
    int s = (e < E) ? src[e] : 0;
    const u16* gsrc = h_bf + (size_t)s * 256 + c;   // 32B per thread
    u16* ldst = &sA[row * 256 + c];
#if USE_ASYNC_LDS
    unsigned loff = (unsigned)(uintptr_t)ldst;      // low 32b of flat addr = LDS offset
    asm volatile("global_load_async_to_lds_b128 %0, %1, off\n\t"
                 "global_load_async_to_lds_b128 %0, %1, off offset:16"
                 :: "v"(loff), "v"(gsrc) : "memory");
    asm volatile("s_wait_asynccnt 0" ::: "memory");
#else
    *(u16x16*)ldst = *(const u16x16*)gsrc;
#endif
  }
  __syncthreads();

  const int lane   = t & 31;
  const int wave   = t >> 5;
  const int mq     = lane & 15;
  const int hiHalf = lane >> 4;
  const int kbA    = hiHalf * 8;
  const int mhalf  = hiHalf * 8;

  // preload all 8 A fragments (64 VGPRs), reused by both column blocks
  bf16x16 afr[8];
#pragma unroll
  for (int s = 0; s < 8; ++s)
    afr[s] = load_a_frag_lds(&sA[mq * 256 + s * 32 + kbA]);

  for (int blk = wave; blk < 16; blk += 8) {
    f32x8 acc = {};
#pragma unroll
    for (int s = 0; s < 8; ++s) {                       // K = 256 in 8 steps of 32
      bf16x16 bfrag = load_b_frag(W1pack + (((size_t)s * 16 + blk) * 32 + lane) * 16);
      acc = __builtin_amdgcn_wmma_f32_16x16x32_bf16(false, afr[s], false, bfrag,
                                                    (short)0, acc, false, false);
    }
    // rank-3 edge-feature update for K = 256..258
    const int i  = blk * 16 + mq;
    const float c0 = W1raw[(size_t)i * 259 + 256];
    const float c1 = W1raw[(size_t)i * 259 + 257];
    const float c2 = W1raw[(size_t)i * 259 + 258];
    if (full) {
#pragma unroll
      for (int r = 0; r < 8; ++r) {
        const int mm = r + mhalf;
        float v = acc[r] + sW[mm * 3 + 0] * c0 + sW[mm * 3 + 1] * c1 + sW[mm * 3 + 2] * c2;
        v = (v > 0.0f) ? v : 0.01f * v;                 // leaky relu
        atomicAdd(hN + (size_t)sDst[mm] * 256 + i, v);  // global_atomic_add_f32
      }
    } else {
#pragma unroll
      for (int r = 0; r < 8; ++r) {
        const int mm = r + mhalf;
        float v = acc[r] + sW[mm * 3 + 0] * c0 + sW[mm * 3 + 1] * c1 + sW[mm * 3 + 2] * c2;
        v = (v > 0.0f) ? v : 0.01f * v;
        if (e0 + mm < E)
          atomicAdd(hN + (size_t)sDst[mm] * 256 + i, v);
      }
    }
  }
}

// ---------------------------------------------------------------------------
// Node kernel: h = relu([h, hN*inv_deg] @ W2^T + b2), h kept in bf16 in place.
// ---------------------------------------------------------------------------
__launch_bounds__(256)
__global__ void node_gemm(u16* __restrict__ h_bf,
                          const float* __restrict__ hN,
                          const float* __restrict__ invd,
                          const u16*  __restrict__ W2pack,  // [16][16][32][16] (this layer)
                          const float* __restrict__ b2,     // [256] (this layer)
                          int N) {
  __shared__ __align__(32) u16 sA[16 * 512];   // 16 nodes x 512 bf16 (concat)
  const int t    = threadIdx.x;
  const int n0   = blockIdx.x * 16;
  const bool full = (n0 + 16 <= N);
  {
    const int row = t >> 4;
    const int c   = (t & 15) * 16;
    int n  = n0 + row;
    int nc = (n < N) ? n : (N - 1);
    *(u16x16*)&sA[row * 512 + c] = *(const u16x16*)(h_bf + (size_t)nc * 256 + c);
    float idg = invd[nc];
    const float* hp = hN + (size_t)nc * 256 + c;
    u16* dp = &sA[row * 512 + 256 + c];
#pragma unroll
    for (int j = 0; j < 16; ++j) dp[j] = f2bf(hp[j] * idg);   // scatter-mean finalize
  }
  __syncthreads();

  const int lane = t & 31, wave = t >> 5;
  const int mq = lane & 15, hiHalf = lane >> 4;
  const int kbA = hiHalf * 8, mhalf = hiHalf * 8;

  for (int blk = wave; blk < 16; blk += 8) {
    f32x8 acc = {};
#pragma unroll
    for (int s = 0; s < 16; ++s) {                      // K = 512 in 16 steps
      bf16x16 afrag = load_a_frag_lds(&sA[mq * 512 + s * 32 + kbA]);
      bf16x16 bfrag = load_b_frag(W2pack + (((size_t)s * 16 + blk) * 32 + lane) * 16);
      acc = __builtin_amdgcn_wmma_f32_16x16x32_bf16(false, afrag, false, bfrag,
                                                    (short)0, acc, false, false);
    }
    const int i = blk * 16 + mq;
    const float bias = b2[i];
    if (full) {
#pragma unroll
      for (int r = 0; r < 8; ++r) {
        const int mm = r + mhalf;
        float v = fmaxf(acc[r] + bias, 0.0f);
        h_bf[(size_t)(n0 + mm) * 256 + i] = f2bf(v);
      }
    } else {
#pragma unroll
      for (int r = 0; r < 8; ++r) {
        const int mm = r + mhalf;
        float v = fmaxf(acc[r] + bias, 0.0f);
        int n = n0 + mm;
        if (n < N) h_bf[(size_t)n * 256 + i] = f2bf(v);
      }
    }
  }
}

// ---------------------------------------------------------------------------
// Head: out = relu(h @ Wh1^T + bh1) @ Wh2^T + bh2
// ---------------------------------------------------------------------------
__launch_bounds__(256)
__global__ void head_kernel(const u16* __restrict__ h_bf,
                            const u16* __restrict__ Wh1pack,   // [8][16][32][16]
                            const float* __restrict__ bh1,
                            const float* __restrict__ Wh2,     // [256]
                            const float* __restrict__ bh2,     // [1]
                            float* __restrict__ out, int N) {
  __shared__ __align__(32) u16 sA[16 * 256];
  __shared__ float sX[16 * 256];
  __shared__ float sP[256];
  const int t  = threadIdx.x;
  const int n0 = blockIdx.x * 16;
  {
    const int row = t >> 4, c = (t & 15) * 16;
    int nc = n0 + row; if (nc >= N) nc = N - 1;
    *(u16x16*)&sA[row * 256 + c] = *(const u16x16*)(h_bf + (size_t)nc * 256 + c);
  }
  __syncthreads();

  const int lane = t & 31, wave = t >> 5;
  const int mq = lane & 15, hiHalf = lane >> 4;
  const int kbA = hiHalf * 8, mhalf = hiHalf * 8;

  for (int blk = wave; blk < 16; blk += 8) {
    f32x8 acc = {};
#pragma unroll
    for (int s = 0; s < 8; ++s) {
      bf16x16 afrag = load_a_frag_lds(&sA[mq * 256 + s * 32 + kbA]);
      bf16x16 bfrag = load_b_frag(Wh1pack + (((size_t)s * 16 + blk) * 32 + lane) * 16);
      acc = __builtin_amdgcn_wmma_f32_16x16x32_bf16(false, afrag, false, bfrag,
                                                    (short)0, acc, false, false);
    }
    const int i = blk * 16 + mq;
    const float bias = bh1[i];
#pragma unroll
    for (int r = 0; r < 8; ++r)
      sX[(r + mhalf) * 256 + i] = fmaxf(acc[r] + bias, 0.0f);
  }
  __syncthreads();

  {
    const int node = t >> 4, seg = t & 15;
    float p = 0.0f;
#pragma unroll
    for (int j = 0; j < 16; ++j)
      p += sX[node * 256 + seg * 16 + j] * Wh2[seg * 16 + j];
    sP[t] = p;
  }
  __syncthreads();
  if (t < 16) {
    int n = n0 + t;
    if (n < N) {
      float s = bh2[0];
#pragma unroll
      for (int j = 0; j < 16; ++j) s += sP[t * 16 + j];
      out[n] = s;
    }
  }
}

// ---------------------------------------------------------------------------
extern "C" void kernel_launch(void* const* d_in, const int* in_sizes, int n_in,
                              void* d_out, int out_size, void* d_ws, size_t ws_size,
                              hipStream_t stream) {
  const int N = in_sizes[0];
  const int E = in_sizes[1];
  const int*   gate = (const int*)d_in[0];
  const int*   src  = (const int*)d_in[1];
  const int*   dst  = (const int*)d_in[2];
  const float* w    = (const float*)d_in[3];
  const float* emb  = (const float*)d_in[4];
  const float* W1   = (const float*)d_in[5];   // [6][256][259]
  const float* W2   = (const float*)d_in[6];   // [6][256][512]
  const float* b2   = (const float*)d_in[7];   // [6][256]
  const float* Wh1  = (const float*)d_in[8];   // [256][256]
  const float* bh1  = (const float*)d_in[9];
  const float* Wh2  = (const float*)d_in[10];
  const float* bh2  = (const float*)d_in[11];
  float* out = (float*)d_out;
  (void)n_in; (void)out_size; (void)ws_size;

  // workspace carve (~157 MB total)
  char* ws = (char*)d_ws;
  auto carve = [&](size_t bytes) { char* p = ws; ws += (bytes + 255) & ~(size_t)255; return p; };
  u16*   h_bf = (u16*)  carve((size_t)N * 256 * 2);
  float* hN   = (float*)carve((size_t)N * 256 * 4);
  float* deg  = (float*)carve((size_t)N * 4);
  float* invd = (float*)carve((size_t)N * 4);
  u16* W1p  = (u16*)carve(6ull * 8  * 16 * 32 * 16 * 2);
  u16* W2p  = (u16*)carve(6ull * 16 * 16 * 32 * 16 * 2);
  u16* Wh1p = (u16*)carve(1ull * 8  * 16 * 32 * 16 * 2);

  // weight pre-pack into WMMA B-fragment order (bf16)
  pack_b_kernel<<<512, 256, 0, stream>>>(W1,  W1p,  6, 8,  259);
  pack_b_kernel<<<512, 256, 0, stream>>>(W2,  W2p,  6, 16, 512);
  pack_b_kernel<<<64,  256, 0, stream>>>(Wh1, Wh1p, 1, 8,  256);

  // h0 = emb[gate_type] (bf16)
  {
    size_t total = (size_t)N * 256;
    init_h_kernel<<<(unsigned)((total + 255) / 256), 256, 0, stream>>>(gate, emb, h_bf, N);
  }

  // degree / inverse degree
  zero_f32_kernel<<<512, 256, 0, stream>>>(deg, (size_t)N);
  degree_kernel<<<(E + 255) / 256, 256, 0, stream>>>(dst, deg, E);
  invdeg_kernel<<<(N + 255) / 256, 256, 0, stream>>>(deg, invd, N);

  const int etiles = (E + 15) / 16;
  const int ntiles = (N + 15) / 16;
  for (int l = 0; l < 6; ++l) {
    zero_f32_kernel<<<2048, 256, 0, stream>>>(hN, (size_t)N * 256);
    edge_gemm_scatter<<<etiles, 256, 0, stream>>>(
        h_bf, src, dst, w,
        W1 + (size_t)l * 256 * 259,
        W1p + (size_t)l * 8 * 16 * 32 * 16,
        hN, E);
    node_gemm<<<ntiles, 256, 0, stream>>>(
        h_bf, hN, invd,
        W2p + (size_t)l * 16 * 16 * 32 * 16,
        b2 + (size_t)l * 256, N);
  }

  head_kernel<<<ntiles, 256, 0, stream>>>(h_bf, Wh1p, bh1, Wh2, bh2, out, N);
}
